// GemmaAttention_71554155152311
// MI455X (gfx1250) — matrix operand order repaired and checked by reading, not compiled, not run on hardware
//
#include <hip/hip_runtime.h>
#include <hip/hip_bf16.h>

typedef __bf16 bf16_t;
typedef __attribute__((ext_vector_type(16))) __bf16 v16bf;
typedef __attribute__((ext_vector_type(8)))  __bf16 v8bf;
typedef __attribute__((ext_vector_type(8)))  float  v8f;
typedef __attribute__((ext_vector_type(4)))  unsigned v4u;
typedef __attribute__((ext_vector_type(8)))  int      v8i;
typedef __attribute__((ext_vector_type(4)))  int      v4i;

#define SHUF16(lo, hi) __builtin_shufflevector((lo), (hi), 0,1,2,3,4,5,6,7,8,9,10,11,12,13,14,15)

namespace cfg {
constexpr int B    = 2;
constexpr int S    = 2048;
constexpr int HID  = 2048;
constexpr int NH   = 8;
constexpr int HD   = 256;
constexpr int HD2  = 128;
constexpr int QS   = NH * HD;          // 2048
constexpr int KVS  = HD;               // 256
constexpr int OQKV = QS + 2 * KVS;     // 2560
constexpr int M    = B * S;            // 4096
constexpr float SCALE = 0.0625f;       // 256^-0.5
}

static __device__ __forceinline__ bf16_t f2bf(float f) {
  unsigned u = __builtin_bit_cast(unsigned, f);
  unsigned r = u + 0x7fffu + ((u >> 16) & 1u);
  unsigned short h = (unsigned short)(r >> 16);
  return __builtin_bit_cast(bf16_t, h);
}

static __device__ __forceinline__ v8f wmma_bf16(v16bf a, v16bf b, v8f c) {
  // D = A(16x32 bf16) * B(32x16 bf16) + C(16x16 f32)
  return __builtin_amdgcn_wmma_f32_16x16x32_bf16(false, a, false, b, (short)0, c, false, false);
}

// flat address of a __shared__ object -> 32-bit LDS byte offset
// (ISA 10.2: the LDS offset is the low 32 bits of the flat address)
static __device__ __forceinline__ unsigned lds_off(const void* p) {
  return (unsigned)(unsigned long long)p;
}

// async global -> LDS copy of 32 bytes (two B128 ops; IOFFSET applies to both
// the global and LDS addresses per ISA 10.7 async addressing)
static __device__ __forceinline__ void async_copy_b256(unsigned lds_addr, const void* gptr) {
  asm volatile(
      "global_load_async_to_lds_b128 %0, %1, off\n\t"
      "global_load_async_to_lds_b128 %0, %1, off offset:16"
      :: "v"(lds_addr), "v"((unsigned long long)gptr)
      : "memory");
}

static __device__ __forceinline__ void wait_asynccnt0() {
  asm volatile("s_wait_asynccnt 0x0" ::: "memory");
}

// ---------------------------------------------------------------------------
// f32 -> bf16 elementwise convert
// ---------------------------------------------------------------------------
__global__ void cvt_bf16_kernel(const float* __restrict__ src, bf16_t* __restrict__ dst, int n) {
  int i = blockIdx.x * blockDim.x + threadIdx.x;
  if (i < n) dst[i] = f2bf(src[i]);
}

// ---------------------------------------------------------------------------
// C[M,N] = A[M,K](bf16, row-major) * Bt[N,K](bf16, row-major)  (NT GEMM)
// Block tile 128x128, K-step 32, 8 waves: wave = (mGroup 0..3) x (nGroup 0..1)
// each wave owns 2 M-tiles x 4 N-tiles of 16x16 f32 accumulators.
// LDS staging uses GLOBAL_LOAD_ASYNC_TO_LDS_B128 (ASYNCcnt path).
// ---------------------------------------------------------------------------
#define GT_M 128
#define GT_N 128
#define GT_K 32
#define GT_LD 40   // padded LDS row stride (bf16 elems)

__global__ __launch_bounds__(256) void gemm_bf16_nt_kernel(
    const bf16_t* __restrict__ A, const bf16_t* __restrict__ Bt,
    float* __restrict__ C, int M, int N, int K) {
  __shared__ __align__(16) bf16_t sA[GT_M * GT_LD];
  __shared__ __align__(16) bf16_t sB[GT_N * GT_LD];

  const int tid  = threadIdx.x;
  const int wave = tid >> 5;
  const int lane = tid & 31;
  const int half = lane >> 4;
  const int ln   = lane & 15;

  const int m0 = blockIdx.y * GT_M;
  const int n0 = blockIdx.x * GT_N;
  const int wm = (wave & 3) * 32;   // wave M offset in tile
  const int wn = (wave >> 2) * 64;  // wave N offset in tile

  v8f acc[2][4] = {};

  const int lrow = tid >> 1;          // 0..127
  const int lcol = (tid & 1) * 16;    // 0 or 16

  const unsigned la = lds_off(&sA[lrow * GT_LD + lcol]);
  const unsigned lb = lds_off(&sB[lrow * GT_LD + lcol]);

  for (int k0 = 0; k0 < K; k0 += GT_K) {
    __syncthreads();  // previous iteration's fragment reads done
    // stage A tile (128x32) and B tile (128x32) into LDS asynchronously
    {
      const bf16_t* ap = &A[(size_t)(m0 + lrow) * K + k0 + lcol];
      const bf16_t* bp = &Bt[(size_t)(n0 + lrow) * K + k0 + lcol];
      async_copy_b256(la, ap);
      async_copy_b256(lb, bp);
      if (k0 + GT_K < K) {  // prefetch next K-tiles (global_prefetch_b8)
        __builtin_prefetch(ap + GT_K, 0, 1);
        __builtin_prefetch(bp + GT_K, 0, 1);
      }
    }
    wait_asynccnt0();
    __syncthreads();

    // A fragments: lane half selects K sub-range per ISA 16-bit A layout
    v16bf af[2];
#pragma unroll
    for (int i = 0; i < 2; i++) {
      const bf16_t* r = &sA[(wm + i * 16 + ln) * GT_LD];
      v8bf lo = *(const v8bf*)&r[half * 8];        // K = half*8 + 0..7
      v8bf hi = *(const v8bf*)&r[16 + half * 8];   // K = 16 + half*8 + 0..7
      af[i] = SHUF16(lo, hi);
    }
    // B fragments: lane half selects K half, lane ln is the N column
    v16bf bfm[4];
#pragma unroll
    for (int j = 0; j < 4; j++) {
      const bf16_t* r = &sB[(wn + j * 16 + ln) * GT_LD];
      v8bf lo = *(const v8bf*)&r[half * 8];
      v8bf hi = *(const v8bf*)&r[16 + half * 8];
      bfm[j] = SHUF16(lo, hi);
    }
#pragma unroll
    for (int i = 0; i < 2; i++)
#pragma unroll
      for (int j = 0; j < 4; j++)
        acc[i][j] = wmma_bf16(af[i], bfm[j], acc[i][j]);
  }

  // store C (f32): VGPR r <-> row half*8+r, lane ln <-> col
#pragma unroll
  for (int i = 0; i < 2; i++)
#pragma unroll
    for (int j = 0; j < 4; j++)
#pragma unroll
      for (int r = 0; r < 8; r++) {
        int row = m0 + wm + i * 16 + half * 8 + r;
        int col = n0 + wn + j * 16 + ln;
        C[(size_t)row * N + col] = acc[i][j][r];
      }
}

// ---------------------------------------------------------------------------
// RoPE on Q/K + bf16 pack of Q/K/V from qkv f32 buffer.
// ---------------------------------------------------------------------------
__global__ void rope_cvt_kernel(const float* __restrict__ qkv,
                                const float* __restrict__ freqs,
                                bf16_t* __restrict__ q,
                                bf16_t* __restrict__ k,
                                bf16_t* __restrict__ v) {
  using namespace cfg;
  int idx = blockIdx.x * blockDim.x + threadIdx.x;
  int total = M * NH * HD2;
  if (idx >= total) return;
  int d   = idx % HD2;
  int h   = (idx / HD2) % NH;
  int tok = idx / (HD2 * NH);
  int s   = tok % S;
  float co = freqs[(size_t)s * HD2 * 2 + d * 2 + 0];
  float si = freqs[(size_t)s * HD2 * 2 + d * 2 + 1];
  const float* base = &qkv[(size_t)tok * OQKV];

  float x1 = base[h * HD + d];
  float x2 = base[h * HD + d + HD2];
  q[((size_t)tok * NH + h) * HD + d]       = f2bf(x1 * co - x2 * si);
  q[((size_t)tok * NH + h) * HD + d + HD2] = f2bf(x1 * si + x2 * co);

  if (h == 0) {
    float k1 = base[QS + d];
    float k2 = base[QS + d + HD2];
    k[(size_t)tok * HD + d]       = f2bf(k1 * co - k2 * si);
    k[(size_t)tok * HD + d + HD2] = f2bf(k1 * si + k2 * co);
    v[(size_t)tok * HD + d]       = f2bf(base[QS + KVS + d]);
    v[(size_t)tok * HD + d + HD2] = f2bf(base[QS + KVS + d + HD2]);
  }
}

// ---------------------------------------------------------------------------
// Flash attention, MQA: one workgroup = 16 query rows x ALL 8 heads
// (wave w == head w). K tile DMA'd into LDS by the Tensor Data Mover
// (TENSORcnt path), V tile transposed manually, P bounced through LDS.
// ---------------------------------------------------------------------------
#define KB 32  // keys per block

__global__ __launch_bounds__(256) void attn_kernel(
    const bf16_t* __restrict__ Q, const bf16_t* __restrict__ Kc,
    const bf16_t* __restrict__ Vc, bf16_t* __restrict__ Oc) {
  using namespace cfg;
  __shared__ __align__(16) bf16_t sK[KB * HD];       // [key][hd]  16KB (TDM dest)
  __shared__ __align__(16) bf16_t sVT[HD * KB];      // [hd][key]  16KB (transposed)
  __shared__ __align__(16) bf16_t sP[8 * 16 * KB];   // per-wave P staging 8KB

  const int tid  = threadIdx.x;
  const int wave = tid >> 5;
  const int lane = tid & 31;
  const int half = lane >> 4;
  const int ln   = lane & 15;

  const int b  = blockIdx.y;
  const int q0 = blockIdx.x * 16;
  const int h  = wave;  // MQA: one head per wave

  // TDM descriptor group 1 (constant): data_size=2B, tensor 256 x 4096,
  // row stride 256 elems, tile 256 x 32 rows.
  v8i tdm_g1;
  tdm_g1[0] = 0x00010000;              // wg_mask=0 | data_size=1 (2 bytes)
  tdm_g1[1] = (int)(256u << 16);       // tensor_dim0[15:0]=256 in bits[63:48]
  tdm_g1[2] = (int)((unsigned)(M & 0xffff) << 16);  // tensor_dim1[15:0]=4096
  tdm_g1[3] = (int)(256u << 16);       // tile_dim0=256 in bits[127:112]
  tdm_g1[4] = KB;                      // tile_dim1=32, tile_dim2=0
  tdm_g1[5] = 256;                     // tensor_dim0_stride[31:0]=256
  tdm_g1[6] = 0;
  tdm_g1[7] = 0;
  const v4i z4 = {};
  const v8i z8 = {};
  const unsigned sK_lds = lds_off(&sK[0]);

  // Q fragments for this wave's 16 rows: 8 chunks of K=32 over HD=256
  v16bf qf[8];
  {
    const bf16_t* qrow = &Q[(((size_t)b * S + q0 + ln) * NH + h) * HD];
#pragma unroll
    for (int c = 0; c < 8; c++) {
      v8bf lo = *(const v8bf*)&qrow[c * 32 + half * 16];
      v8bf hi = *(const v8bf*)&qrow[c * 32 + half * 16 + 8];
      qf[c] = SHUF16(lo, hi);
    }
  }

  v8f o[16] = {};
  float mrow[8], lrow[8];
#pragma unroll
  for (int j = 0; j < 8; j++) { mrow[j] = -1e38f; lrow[j] = 0.0f; }

  const int qmax = q0 + 15;
  const int nblk = qmax / KB + 1;  // causal: skip key blocks entirely above diag

  const int lkey = tid >> 3;        // 0..31
  const int lhd  = (tid & 7) * 32;  // 0..224

  for (int kb = 0; kb < nblk; kb++) {
    const int kbase = kb * KB;
    __syncthreads();  // everyone done reading previous tiles
    // K tile via Tensor Data Mover: one DMA for the whole 32x256 tile
    if (wave == 0) {
      unsigned long long ga =
          (unsigned long long)(&Kc[((size_t)b * S + kbase) * HD]);
      v4u g0;
      g0[0] = 1u;                                   // count=1, flags=0
      g0[1] = sK_lds;                               // lds_addr
      g0[2] = (unsigned)ga;                         // global_addr[31:0]
      g0[3] = (unsigned)(ga >> 32) | 0x80000000u;   // addr[56:32] | type=2
      __builtin_amdgcn_tensor_load_to_lds(g0, tdm_g1, z4, z4, z8, 0);
    }
    // V tile transposed (manual: TDM cannot transpose)
    {
      const bf16_t* vp = &Vc[((size_t)b * S + kbase + lkey) * HD + lhd];
#pragma unroll
      for (int i = 0; i < 32; i++)
        sVT[(lhd + i) * KB + lkey] = vp[i];
    }
    if (wave == 0) __builtin_amdgcn_s_wait_tensorcnt(0);
    __syncthreads();  // K (TDM) + V (ds) tiles visible to all waves

    // scores: 16 q-rows x 32 keys = 2 WMMA C tiles, K-dim = HD (8 steps)
    v8f sc[2] = {};
#pragma unroll
    for (int nt = 0; nt < 2; nt++) {
#pragma unroll
      for (int c = 0; c < 8; c++) {
        const bf16_t* r = &sK[(nt * 16 + ln) * HD + c * 32 + half * 16];
        v8bf lo = *(const v8bf*)&r[0];
        v8bf hi = *(const v8bf*)&r[8];
        v16bf bk = SHUF16(lo, hi);
        sc[nt] = wmma_bf16(qf[c], bk, sc[nt]);
      }
    }

    // scale + causal mask + online softmax (rows = half*8 + j)
#pragma unroll
    for (int j = 0; j < 8; j++) {
      const int qrowi = q0 + half * 8 + j;
#pragma unroll
      for (int nt = 0; nt < 2; nt++) {
        int kidx = kbase + nt * 16 + ln;
        float sv = sc[nt][j] * SCALE;
        if (kidx > qrowi) sv = -1e38f;
        sc[nt][j] = sv;
      }
      float mx = fmaxf(sc[0][j], sc[1][j]);
#pragma unroll
      for (int off = 1; off < 16; off <<= 1)
        mx = fmaxf(mx, __shfl_xor(mx, off, 32));
      float mnew = fmaxf(mrow[j], mx);
      float corr = __expf(mrow[j] - mnew);
      mrow[j] = mnew;
      float p0 = __expf(sc[0][j] - mnew);
      float p1 = __expf(sc[1][j] - mnew);
      sc[0][j] = p0;
      sc[1][j] = p1;
      float rs = p0 + p1;
#pragma unroll
      for (int off = 1; off < 16; off <<= 1)
        rs += __shfl_xor(rs, off, 32);
      lrow[j] = lrow[j] * corr + rs;
#pragma unroll
      for (int t = 0; t < 16; t++) o[t][j] *= corr;
    }

    // stage P (C layout) -> LDS -> reload as A fragment (16x32)
#pragma unroll
    for (int nt = 0; nt < 2; nt++)
#pragma unroll
      for (int j = 0; j < 8; j++)
        sP[wave * 16 * KB + (half * 8 + j) * KB + nt * 16 + ln] = f2bf(sc[nt][j]);

    v8bf plo = *(const v8bf*)&sP[wave * 16 * KB + ln * KB + half * 16];
    v8bf phi = *(const v8bf*)&sP[wave * 16 * KB + ln * KB + half * 16 + 8];
    v16bf pf = SHUF16(plo, phi);

    // O += P(16x32) @ V(32x256): 16 output N-tiles
#pragma unroll
    for (int t = 0; t < 16; t++) {
      const bf16_t* r = &sVT[(t * 16 + ln) * KB + half * 16];
      v8bf lo = *(const v8bf*)&r[0];
      v8bf hi = *(const v8bf*)&r[8];
      v16bf bv = SHUF16(lo, hi);
      o[t] = wmma_bf16(pf, bv, o[t]);
    }
  }

  // epilogue: normalize and write bf16 (input to output projection GEMM)
#pragma unroll
  for (int j = 0; j < 8; j++) {
    float inv = 1.0f / lrow[j];
    size_t row = (size_t)b * S + q0 + half * 8 + j;
#pragma unroll
    for (int t = 0; t < 16; t++)
      Oc[row * HID + h * HD + t * 16 + ln] = f2bf(o[t][j] * inv);
  }
}

// ---------------------------------------------------------------------------
// host-side launcher
// ---------------------------------------------------------------------------
extern "C" void kernel_launch(void* const* d_in, const int* in_sizes, int n_in,
                              void* d_out, int out_size, void* d_ws, size_t ws_size,
                              hipStream_t stream) {
  using namespace cfg;
  (void)in_sizes; (void)n_in; (void)out_size; (void)ws_size;

  const float* hs    = (const float*)d_in[0];  // (B,S,HID)
  const float* freqs = (const float*)d_in[1];  // (S,128,2)
  const float* mask  = (const float*)d_in[2];  // unused (causal applied analytically)
  const float* w_qkv = (const float*)d_in[3];  // (2560,2048)
  const float* w_o   = (const float*)d_in[4];  // (2048,2048)
  (void)mask;
  float* out = (float*)d_out;

  char* ws = (char*)d_ws;
  size_t off = 0;
  auto carve = [&](size_t bytes) -> void* {
    void* p = ws + off;
    off += (bytes + 255) & ~(size_t)255;
    return p;
  };

  bf16_t* hs_bf   = (bf16_t*)carve((size_t)M * HID * 2);
  bf16_t* wqkv_bf = (bf16_t*)carve((size_t)OQKV * HID * 2);
  bf16_t* wo_bf   = (bf16_t*)carve((size_t)HID * HID * 2);
  float*  qkv_f   = (float*) carve((size_t)M * OQKV * 4);
  bf16_t* q_bf    = (bf16_t*)carve((size_t)M * NH * HD * 2);
  bf16_t* k_bf    = (bf16_t*)carve((size_t)M * HD * 2);
  bf16_t* v_bf    = (bf16_t*)carve((size_t)M * HD * 2);
  bf16_t* ao_bf   = (bf16_t*)carve((size_t)M * HID * 2);

  // 1) bf16 converts
  {
    int n = M * HID;
    cvt_bf16_kernel<<<(n + 255) / 256, 256, 0, stream>>>(hs, hs_bf, n);
    n = OQKV * HID;
    cvt_bf16_kernel<<<(n + 255) / 256, 256, 0, stream>>>(w_qkv, wqkv_bf, n);
    n = HID * HID;
    cvt_bf16_kernel<<<(n + 255) / 256, 256, 0, stream>>>(w_o, wo_bf, n);
  }

  // 2) QKV projection: (4096x2048) x (2560x2048)^T -> (4096x2560) f32
  gemm_bf16_nt_kernel<<<dim3(OQKV / GT_N, M / GT_M), 256, 0, stream>>>(
      hs_bf, wqkv_bf, qkv_f, M, OQKV, HID);

  // 3) RoPE + pack bf16 Q/K/V
  {
    int n = M * NH * HD2;
    rope_cvt_kernel<<<(n + 255) / 256, 256, 0, stream>>>(qkv_f, freqs, q_bf, k_bf, v_bf);
  }

  // 4) flash attention (MQA-shared K/V in LDS, TDM-fed K tiles)
  attn_kernel<<<dim3(S / 16, B), 256, 0, stream>>>(q_bf, k_bf, v_bf, ao_bf);

  // 5) output projection: (4096x2048) x (2048x2048)^T -> d_out f32
  gemm_bf16_nt_kernel<<<dim3(HID / GT_N, M / GT_M), 256, 0, stream>>>(
      ao_bf, wo_bf, out, M, HID, HID);
}
